// GNN_node_test_71794673320191
// MI455X (gfx1250) — compile-verified, matrix operand order checked by reading, not run on hardware
//
#include <hip/hip_runtime.h>

#define N_NODES 50000
#define N_EDGES 300000
#define IN_DIM  64
#define EMB     256
#define L_OUT   3
#define L_IN    3
#define BN_EPS  1e-5f

typedef __attribute__((ext_vector_type(16))) __bf16          v16bf;
typedef __attribute__((ext_vector_type(16))) unsigned short  v16us;
typedef __attribute__((ext_vector_type(8)))  unsigned short  v8us;
typedef __attribute__((ext_vector_type(8)))  float           v8f;

// fp32 -> bf16 bits, round-to-nearest-even
__device__ __forceinline__ unsigned short f2bf(float f) {
  unsigned int u = __float_as_uint(f);
  unsigned int r = u + 0x7fffu + ((u >> 16) & 1u);
  return (unsigned short)(r >> 16);
}

// ---------------------------------------------------------------------------
// Fused GEMM: out[M x 256] = act( (A (+ A2)) [M x K] @ W [K x 256] + bias )
// W supplied transposed as bf16 bits: Wt[n][k], row-major [256][K].
//
// Block = 512 threads = 16 waves; one block owns a 16-row x 256-col output
// stripe. The 16 x K fp32 A-stripe (with fused +A2) is converted to bf16 ONCE
// and staged in LDS; each wave computes one 16x16 tile via
// v_wmma_f32_16x16x32_bf16, pulling A-fragments from LDS (ds_load_b128) and
// B-fragments from L2-resident global bf16 weights (global_load_b128).
// M multiple of 16 (50000 = 16*3125). K in {64, 256}.
// ---------------------------------------------------------------------------
__global__ void __launch_bounds__(512)
wmma_gemm_bias_act(const float* __restrict__ A,
                   const float* __restrict__ A2,            // nullable: fused z = A + A2
                   const unsigned short* __restrict__ Wt,   // bf16 bits, [256][K]
                   const float* __restrict__ bias,          // [256]
                   float* __restrict__ out,                 // [M x 256]
                   int K, int relu)
{
  __shared__ unsigned short smem[16 * EMB];    // 8 KB max (K=256)

  const int tm       = blockIdx.x;             // row tile (0..3124)
  const int rowStart = tm * 16;

  // ---- cooperative stage: A stripe (16 x K fp32) -> bf16 in LDS ----
  {
    const int ept  = (16 * K) >> 9;            // elements per thread: K/32 (8 or 2)
    const int base = (int)threadIdx.x * ept;   // within [0, 16*K)
    const int row  = base / K;                 // no row straddle: 32 threads/row
    const int k    = base - row * K;
    const float* gsrc  = A + (long)(rowStart + row) * K + k;
    const float* gsrc2 = A2 ? (A2 + (long)(rowStart + row) * K + k) : nullptr;
    for (int j = 0; j < ept; ++j) {
      float v = gsrc[j];
      if (gsrc2) v += gsrc2[j];
      smem[base + j] = f2bf(v);
    }
  }
  __syncthreads();

  const int wave = (int)(threadIdx.x >> 5);    // 0..15 == column tile tn
  const int lane = (int)(threadIdx.x & 31u);
  const int idx  = lane & 15;
  const int half = lane >> 4;

  const int ncol = wave * 16 + idx;            // output column this lane owns
  const unsigned short* bcol = Wt + (long)ncol * K;
  const unsigned short* lrow = smem + idx * K; // A row this lane feeds

  v8f acc = {};
  for (int kb = 0; kb < K; kb += 32) {
    const int k0 = kb + 8 * half;   // ISA A/B layout: K = kb + 8*half + e (e<8),
                                    //                 K = kb + 16 + 8*half + (e-8)
    // ---- A fragment: two ds_load_b128 (8 bf16 each) from LDS ----
    const v8us a0 = *(const v8us*)(lrow + k0);
    const v8us a1 = *(const v8us*)(lrow + k0 + 16);
    // ---- B fragment: two global b128 loads of bf16 bits ----
    const v8us b0 = *(const v8us*)(bcol + k0);
    const v8us b1 = *(const v8us*)(bcol + k0 + 16);

    v16us au, bu;
    #pragma unroll
    for (int e = 0; e < 8; ++e) {
      au[e] = a0[e]; au[e + 8] = a1[e];
      bu[e] = b0[e]; bu[e + 8] = b1[e];
    }
    const v16bf av = __builtin_bit_cast(v16bf, au);
    const v16bf bv = __builtin_bit_cast(v16bf, bu);
    acc = __builtin_amdgcn_wmma_f32_16x16x32_bf16(
        /*neg_a=*/false, av, /*neg_b=*/false, bv,
        /*c_mod=*/(short)0, acc, /*reuse_a=*/false, /*reuse_b=*/false);
  }

  const float bb = bias[ncol];
  #pragma unroll
  for (int r = 0; r < 8; ++r) {
    float v = acc[r] + bb;
    if (relu) v = fmaxf(v, 0.0f);
    out[(long)(rowStart + r + 8 * half) * EMB + ncol] = v;
  }
}

// ---------------------------------------------------------------------------
// Weight conversion: W [mats][K][N] fp32 -> Wt [mats][N][K] bf16 bits
// ---------------------------------------------------------------------------
__global__ void __launch_bounds__(256)
convert_transpose(const float* __restrict__ W, unsigned short* __restrict__ Wt,
                  int K, int N, long total)
{
  long t = blockIdx.x * (long)blockDim.x + threadIdx.x;
  if (t >= total) return;
  const long kn = (long)K * N;
  const long m  = t / kn;
  const long r  = t - m * kn;
  const int  n  = (int)(r / K);
  const int  k  = (int)(r - (long)n * K);
  Wt[t] = f2bf(W[m * kn + (long)k * N + n]);
}

__global__ void __launch_bounds__(256)
zero_f4(float* __restrict__ p, long n4)   // zeros n4 float4's
{
  long t = blockIdx.x * (long)blockDim.x + threadIdx.x;
  if (t < n4) ((float4*)p)[t] = make_float4(0.f, 0.f, 0.f, 0.f);
}

// agg[dst] += g[src], float4 chunks, global_atomic_add_f32
__global__ void __launch_bounds__(256)
scatter_add(const float* __restrict__ g, const int* __restrict__ src,
            const int* __restrict__ dst, float* __restrict__ agg)
{
  const long t = blockIdx.x * (long)blockDim.x + threadIdx.x;
  const long e = t >> 6;                 // 64 float4-chunks per edge (256 feats)
  if (e >= N_EDGES) return;
  const int c = (int)(t & 63) << 2;
  const int s = src[e];
  const int d = dst[e];
  const float4 v = *(const float4*)(g + (long)s * EMB + c);
  float* p = agg + (long)d * EMB + c;
  atomicAdd(p + 0, v.x);
  atomicAdd(p + 1, v.y);
  atomicAdd(p + 2, v.z);
  atomicAdd(p + 3, v.w);
}

// per-channel sum / sumsq (stats[0..255]=sum, stats[256..511]=sumsq)
__global__ void __launch_bounds__(256)
bn_stats(const float* __restrict__ g, float* __restrict__ stats)
{
  const int c = threadIdx.x;             // channel
  float s = 0.f, ss = 0.f;
  for (int r = blockIdx.x; r < N_NODES; r += gridDim.x) {
    const float v = g[(long)r * EMB + c];
    s += v; ss += v * v;
  }
  atomicAdd(&stats[c], s);
  atomicAdd(&stats[EMB + c], ss);
}

__global__ void __launch_bounds__(256)
bn_apply(const float* __restrict__ g, const float* __restrict__ stats,
         const float* __restrict__ gamma, const float* __restrict__ beta,
         float* __restrict__ out, int relu)
{
  const long t = blockIdx.x * (long)blockDim.x + threadIdx.x;
  if (t >= (long)N_NODES * EMB) return;
  const int c = (int)(t & (EMB - 1));
  const float inv_n = 1.0f / (float)N_NODES;
  const float mean  = stats[c] * inv_n;
  const float var   = stats[EMB + c] * inv_n - mean * mean;
  float v = (g[t] - mean) * rsqrtf(var + BN_EPS) * gamma[c] + beta[c];
  if (relu) v = fmaxf(v, 0.0f);
  out[t] = v;
}

// ---------------------------------------------------------------------------
extern "C" void kernel_launch(void* const* d_in, const int* in_sizes, int n_in,
                              void* d_out, int out_size, void* d_ws, size_t ws_size,
                              hipStream_t stream) {
  (void)in_sizes; (void)n_in; (void)out_size; (void)ws_size;

  const float* x     = (const float*)d_in[0];   // [50000 x 64]
  const int*   ei    = (const int*)  d_in[1];   // [2 x 300000]
  const float* Wenc  = (const float*)d_in[2];   // [64 x 256]
  const float* benc  = (const float*)d_in[3];   // [256]
  const float* W1    = (const float*)d_in[4];   // [3][3][256][256]
  const float* b1    = (const float*)d_in[5];   // [3][3][256]
  const float* W2    = (const float*)d_in[6];
  const float* b2    = (const float*)d_in[7];
  const float* gamma = (const float*)d_in[8];   // [3][256]
  const float* beta  = (const float*)d_in[9];

  const int* src = ei;
  const int* dst = ei + N_EDGES;

  // ---- workspace carve-up ----
  const long FE = (long)N_NODES * EMB;          // 12.8M floats per feature buffer
  char* ws = (char*)d_ws;
  float* h    = (float*)ws;                         ws += FE * sizeof(float);
  float* agg  = (float*)ws;                         ws += FE * sizeof(float);
  float* tmp  = (float*)ws;                         ws += FE * sizeof(float);
  unsigned short* wenc_t = (unsigned short*)ws;     ws += (long)IN_DIM * EMB * sizeof(unsigned short);
  unsigned short* w1_t   = (unsigned short*)ws;     ws += 9L * EMB * EMB * sizeof(unsigned short);
  unsigned short* w2_t   = (unsigned short*)ws;     ws += 9L * EMB * EMB * sizeof(unsigned short);
  float* stats = (float*)ws;                        ws += 2 * EMB * sizeof(float);

  // ---- weight conversion (fp32 -> transposed bf16) ----
  {
    long t0 = (long)IN_DIM * EMB;
    convert_transpose<<<(int)((t0 + 255) / 256), 256, 0, stream>>>(Wenc, wenc_t, IN_DIM, EMB, t0);
    long t1 = 9L * EMB * EMB;
    convert_transpose<<<(int)((t1 + 255) / 256), 256, 0, stream>>>(W1, w1_t, EMB, EMB, t1);
    convert_transpose<<<(int)((t1 + 255) / 256), 256, 0, stream>>>(W2, w2_t, EMB, EMB, t1);
  }

  const int gemm_blocks = N_NODES / 16;             // 3125 (16 waves/block)

  // ---- encoder: h = x @ Wenc + benc ----
  wmma_gemm_bias_act<<<gemm_blocks, 512, 0, stream>>>(x, nullptr, wenc_t, benc, h,
                                                      IN_DIM, /*relu=*/0);

  const int zero_blocks    = (int)((FE / 4 + 255) / 256);
  const int scatter_blocks = (int)(((long)N_EDGES * 64 + 255) / 256);
  const int elem_blocks    = (int)((FE + 255) / 256);

  for (int l = 0; l < L_OUT; ++l) {
    for (int i = 0; i < L_IN; ++i) {
      const int li = l * L_IN + i;
      zero_f4<<<zero_blocks, 256, 0, stream>>>(agg, FE / 4);
      scatter_add<<<scatter_blocks, 256, 0, stream>>>(h, src, dst, agg);
      // tmp = relu( (h + agg) @ W1 + b1 )
      wmma_gemm_bias_act<<<gemm_blocks, 512, 0, stream>>>(
          h, agg, w1_t + (long)li * EMB * EMB, b1 + (long)li * EMB, tmp, EMB, /*relu=*/1);
      // h = act( tmp @ W2 + b2 ), inner ReLU except after last inner conv
      wmma_gemm_bias_act<<<gemm_blocks, 512, 0, stream>>>(
          tmp, nullptr, w2_t + (long)li * EMB * EMB, b2 + (long)li * EMB, h, EMB,
          /*relu=*/(i < L_IN - 1) ? 1 : 0);
    }
    // batch norm over nodes, then ReLU except after last outer layer
    zero_f4<<<1, 256, 0, stream>>>(stats, (2 * EMB) / 4);
    bn_stats<<<256, 256, 0, stream>>>(h, stats);
    float* dest = (l == L_OUT - 1) ? (float*)d_out : h;
    bn_apply<<<elem_blocks, 256, 0, stream>>>(h, stats, gamma + (long)l * EMB,
                                              beta + (long)l * EMB, dest,
                                              /*relu=*/(l < L_OUT - 1) ? 1 : 0);
  }
}
